// EdgeModel_7060926234898
// MI455X (gfx1250) — matrix operand order repaired and checked by reading, not compiled, not run on hardware
//
#include <hip/hip_runtime.h>
#include <hip/hip_bf16.h>

typedef __attribute__((ext_vector_type(16))) __bf16 v16bf;
typedef __attribute__((ext_vector_type(8)))  __bf16 v8bf;
typedef __attribute__((ext_vector_type(8)))  float  v8f;

#define IN_DIM 192
#define HID    64
#define OUTD   64
#define S1     72   // LDS row stride (halves) for W1, padded vs 64 to break bank conflicts
#define S2     72
#define HS     72   // per-wave h staging row stride (halves)
#define WAVES  8

struct BPair { v8bf lo, hi; };

static __device__ inline v16bf pack16(v8bf lo, v8bf hi) {
    BPair p{lo, hi};
    return __builtin_bit_cast(v16bf, p);
}

// feature k in [0,192): 0-63 -> src, 64-127 -> dest, 128-191 -> edge_attr
static __device__ inline const float* fsel(const float* __restrict__ s,
                                           const float* __restrict__ d,
                                           const float* __restrict__ e,
                                           long edge, int k) {
    const float* base = (k < 64) ? s : ((k < 128) ? d : e);
    return base + edge * 64 + (k & 63);
}

// load 8 contiguous f32 (two b128 loads) and convert to 8 bf16
static __device__ inline v8bf cvt8(const float* __restrict__ p) {
    const float4 a = ((const float4*)p)[0];
    const float4 b = ((const float4*)p)[1];
    v8bf r;
    r[0] = (__bf16)a.x; r[1] = (__bf16)a.y; r[2] = (__bf16)a.z; r[3] = (__bf16)a.w;
    r[4] = (__bf16)b.x; r[5] = (__bf16)b.y; r[6] = (__bf16)b.z; r[7] = (__bf16)b.w;
    return r;
}

__global__ __launch_bounds__(256)
void edge_mlp_wmma(const float* __restrict__ src, const float* __restrict__ dst,
                   const float* __restrict__ ea,
                   const float* __restrict__ W1, const float* __restrict__ B1,
                   const float* __restrict__ W2, const float* __restrict__ B2,
                   float* __restrict__ out, int E) {
    __shared__ __bf16 w1s[IN_DIM * S1];
    __shared__ __bf16 w2s[HID * S2];
    __shared__ float  b1s[HID];
    __shared__ float  b2s[OUTD];
    __shared__ __bf16 hstage[WAVES][16 * HS];

    const int tid = threadIdx.x;

    // ---- stage weights (bf16) + biases (f32) into LDS, once per block ----
    for (int i = tid; i < IN_DIM * HID; i += 256)
        w1s[(i >> 6) * S1 + (i & 63)] = (__bf16)W1[i];
    for (int i = tid; i < HID * OUTD; i += 256)
        w2s[(i >> 6) * S2 + (i & 63)] = (__bf16)W2[i];
    if (tid < HID)                 b1s[tid]       = B1[tid];
    else if (tid < HID + OUTD)     b2s[tid - HID] = B2[tid - HID];
    __syncthreads();

    const int lane = tid & 31;
    const int wave = tid >> 5;
    const int m    = lane & 15;   // A-layout row / D-layout column this lane owns
    const int hi   = lane >> 4;   // half-wave selector

    __bf16* hw = hstage[wave];

    const long tiles = (long)E >> 4;           // 16 edges per tile
    for (long t = (long)blockIdx.x * WAVES + wave; t < tiles;
         t += (long)gridDim.x * WAVES) {
        const long ebase = t << 4;

        // ---- layer 1: acc = bias1, then 6 k-steps x 4 n-tiles of WMMA ----
        v8f acc1[4];
        #pragma unroll
        for (int nt = 0; nt < 4; ++nt) {
            float bv = b1s[m + nt * 16];
            acc1[nt] = (v8f){bv, bv, bv, bv, bv, bv, bv, bv};
        }

        #pragma unroll
        for (int ks = 0; ks < 6; ++ks) {
            // A fragment: lane holds row m, halves 0..7 = K kA..kA+7, 8..15 = K kB..kB+7
            const int kA = ks * 32 + hi * 8;
            const int kB = kA + 16;
            const v16bf afrag = pack16(cvt8(fsel(src, dst, ea, ebase + m, kA)),
                                       cvt8(fsel(src, dst, ea, ebase + m, kB)));
            const int krow = ks * 32 + hi * 8;   // B fragment: lane = column N, elements over K (table order)
            #pragma unroll
            for (int nt = 0; nt < 4; ++nt) {
                v16bf bfrag;
                #pragma unroll
                for (int s = 0; s < 16; ++s)
                    bfrag[s] = w1s[(krow + (s & 7) + (s >> 3) * 16) * S1 + nt * 16 + m];
                acc1[nt] = __builtin_amdgcn_wmma_f32_16x16x32_bf16(
                    false, afrag, false, bfrag, (short)0, acc1[nt], false, false);
            }
        }

        // ---- ReLU, convert to bf16, stage through wave-private LDS ----
        // (D layout: lane owns column N = m + nt*16, rows r + hi*8)
        #pragma unroll
        for (int nt = 0; nt < 4; ++nt) {
            #pragma unroll
            for (int r = 0; r < 8; ++r) {
                float v = acc1[nt][r];
                v = v > 0.f ? v : 0.f;
                hw[(r + hi * 8) * HS + m + nt * 16] = (__bf16)v;
            }
        }
        // DS ops from one wave are processed in order; drain before re-reading
        asm volatile("s_wait_dscnt 0" ::: "memory");

        // ---- layer 2: acc = bias2, 2 k-steps x 4 n-tiles ----
        v8f acc2[4];
        #pragma unroll
        for (int nt = 0; nt < 4; ++nt) {
            float bv = b2s[m + nt * 16];
            acc2[nt] = (v8f){bv, bv, bv, bv, bv, bv, bv, bv};
        }

        #pragma unroll
        for (int ks = 0; ks < 2; ++ks) {
            const int kA = ks * 32 + hi * 8;
            const int kB = kA + 16;
            const v16bf afrag = pack16(*(const v8bf*)&hw[m * HS + kA],
                                       *(const v8bf*)&hw[m * HS + kB]);
            const int krow = ks * 32 + hi * 8;
            #pragma unroll
            for (int nt = 0; nt < 4; ++nt) {
                v16bf bfrag;
                #pragma unroll
                for (int s = 0; s < 16; ++s)
                    bfrag[s] = w2s[(krow + (s & 7) + (s >> 3) * 16) * S2 + nt * 16 + m];
                acc2[nt] = __builtin_amdgcn_wmma_f32_16x16x32_bf16(
                    false, afrag, false, bfrag, (short)0, acc2[nt], false, false);
            }
        }

        // ---- store f32 output: half-wave writes 64B contiguous per row ----
        float* obase = out + ebase * 64;
        #pragma unroll
        for (int nt = 0; nt < 4; ++nt) {
            #pragma unroll
            for (int r = 0; r < 8; ++r)
                obase[(r + hi * 8) * 64 + m + nt * 16] = acc2[nt][r];
        }
    }
}

extern "C" void kernel_launch(void* const* d_in, const int* in_sizes, int n_in,
                              void* d_out, int out_size, void* d_ws, size_t ws_size,
                              hipStream_t stream) {
    const float* src = (const float*)d_in[0];
    const float* dst = (const float*)d_in[1];
    const float* ea  = (const float*)d_in[2];
    // d_in[3] = u, d_in[4] = batch : unused by the reference computation
    const float* W1  = (const float*)d_in[5];
    const float* B1  = (const float*)d_in[6];
    const float* W2  = (const float*)d_in[7];
    const float* B2  = (const float*)d_in[8];
    float* out = (float*)d_out;

    const int E = in_sizes[0] / 64;          // 1,048,576
    const long tiles = (long)E / 16;         // 65,536 tiles of 16 edges
    long blocks = (tiles + WAVES - 1) / WAVES;
    if (blocks > 2048) blocks = 2048;        // grid-stride: ~4 tiles per wave

    hipLaunchKernelGGL(edge_mlp_wmma, dim3((unsigned)blocks), dim3(256), 0, stream,
                       src, dst, ea, W1, B1, W2, B2, out, E);
}
